// RoleScorer_85392539779535
// MI455X (gfx1250) — matrix operand order repaired and checked by reading, not compiled, not run on hardware
//
#include <hip/hip_runtime.h>

// ---------------------------------------------------------------------------
// Problem constants (reference: B,S,D,H,O = 4,128,768,768,2)
// ---------------------------------------------------------------------------
#define BB 4
#define SS 128
#define DD 768
#define HH 768
#define OO 2

typedef __attribute__((ext_vector_type(16))) __bf16 v16bf;
typedef __attribute__((ext_vector_type(8)))  __bf16 v8bf;
typedef __attribute__((ext_vector_type(8)))  float  v8f;
typedef __attribute__((ext_vector_type(4)))  int    v4i;

typedef __attribute__((address_space(1))) v4i* gv4i_p;   // global int4*
typedef __attribute__((address_space(3))) v4i* lv4i_p;   // LDS int4*

__device__ __forceinline__ __bf16 relu_bf(float x) {
    return (__bf16)fmaxf(x, 0.0f);
}

#if __has_builtin(__builtin_amdgcn_global_load_async_to_lds_b128)
#define HAVE_ASYNC_LDS 1
#else
#define HAVE_ASYNC_LDS 0
#endif

__device__ __forceinline__ void wait_asynccnt0() {
#if __has_builtin(__builtin_amdgcn_s_wait_asynccnt)
    __builtin_amdgcn_s_wait_asynccnt(0);
#else
    asm volatile("s_wait_asynccnt 0x0" ::: "memory");
#endif
}

// ---------------------------------------------------------------------------
// Kernel 1: precision/layout prep
//   a,b (f32 row-major)      -> bf16 row-major
//   W1 (1536x768 f32)        -> two 768x768 bf16 transposed halves (row = h, K-contig)
//   W2 (768x2 f32)           -> 16x768 bf16 padded transpose (rows 0..1 = cols of W2)
// ---------------------------------------------------------------------------
__global__ void prep_kernel(const float* __restrict__ a,
                            const float* __restrict__ b,
                            const float* __restrict__ W1,
                            const float* __restrict__ W2,
                            __bf16* __restrict__ a_bf,
                            __bf16* __restrict__ b_bf,
                            __bf16* __restrict__ w1aT,
                            __bf16* __restrict__ w1bT,
                            __bf16* __restrict__ w2t)
{
    const int NA  = BB * SS * DD;   // 393216
    const int NW  = HH * DD;        // 589824
    const int NW2 = 16 * HH;        // 12288
    const int tid    = blockIdx.x * blockDim.x + threadIdx.x;
    const int stride = gridDim.x * blockDim.x;

    for (int i = tid; i < NA; i += stride) {
        a_bf[i] = (__bf16)a[i];
        b_bf[i] = (__bf16)b[i];
    }
    for (int i = tid; i < NW; i += stride) {
        const int h = i / DD;
        const int d = i - h * DD;
        w1aT[i] = (__bf16)W1[d * HH + h];          // W1[:D] transposed
        w1bT[i] = (__bf16)W1[(d + DD) * HH + h];   // W1[D:] transposed
    }
    for (int i = tid; i < NW2; i += stride) {
        const int o = i / HH;
        const int h = i - o * HH;
        w2t[i] = (o < OO) ? (__bf16)W2[h * OO + o] : (__bf16)0.0f;
    }
}

// ---------------------------------------------------------------------------
// Kernel 2: both projection GEMMs in one launch.
//   waves [0, 768)   : ha[m,h] = sum_d a[m,d]*W1aT[h,d] + b1[h]
//   waves [768, 1536): hb[m,h] = sum_d b[m,d]*W1bT[h,d]
// One wave computes a 16x32 output tile (2 WMMAs / K-step); 1536 waves total
// for latency hiding (instruction-level stalls can't be hidden at 1 wave/SIMD).
// ---------------------------------------------------------------------------
__global__ void proj_gemm_kernel(const __bf16* __restrict__ a_bf,
                                 const __bf16* __restrict__ b_bf,
                                 const __bf16* __restrict__ w1aT,
                                 const __bf16* __restrict__ w1bT,
                                 const float*  __restrict__ b1,
                                 float* __restrict__ haBuf,
                                 float* __restrict__ hbBuf)
{
    const int lane = threadIdx.x & 31;
    const int gw   = blockIdx.x * (blockDim.x >> 5) + (threadIdx.x >> 5);
    const int sel  = gw >= 768;               // 0: ha path, 1: hb path
    const int w    = sel ? (gw - 768) : gw;
    const int NT2  = HH / 32;                 // 24 N-supertiles of 32
    const int m0   = (w / NT2) * 16;          // M = 512 -> 32 tiles
    const int n0   = (w % NT2) * 32;
    const int half = lane >> 4;               // 0: lanes 0-15, 1: lanes 16-31
    const int lr   = lane & 15;

    const __bf16* A  = sel ? b_bf : a_bf;
    const __bf16* WT = sel ? w1bT : w1aT;
    float*        C  = sel ? hbBuf : haBuf;

    const __bf16* arow  = A  + (size_t)(m0 + lr) * DD;
    const __bf16* wrow0 = WT + (size_t)(n0 +  0 + lr) * DD;
    const __bf16* wrow1 = WT + (size_t)(n0 + 16 + lr) * DD;

    v8f acc0 = {}, acc1 = {};

    for (int k0 = 0; k0 < DD; k0 += 32) {
        __builtin_prefetch((const void*)(arow + k0 + 128), 0, 3);

        // A fragment 16x32: lanes 0-15 hold K=[k0..k0+15],
        //                   lanes 16-31 shifted by 16.
        const int ka = k0 + half * 16;
        v8bf alo = *(const v8bf*)(arow + ka);
        v8bf ahi = *(const v8bf*)(arow + ka + 8);
        v16bf af;
        #pragma unroll
        for (int i = 0; i < 8; ++i) { af[i] = alo[i]; af[i + 8] = ahi[i]; }

        // B fragments 32x16: lane = column n, K contiguous per half-wave.
        const int kb = k0 + half * 16;
        v8bf b0lo = *(const v8bf*)(wrow0 + kb);
        v8bf b0hi = *(const v8bf*)(wrow0 + kb + 8);
        v8bf b1lo = *(const v8bf*)(wrow1 + kb);
        v8bf b1hi = *(const v8bf*)(wrow1 + kb + 8);
        v16bf bf0, bf1;
        #pragma unroll
        for (int i = 0; i < 8; ++i) {
            bf0[i] = b0lo[i]; bf0[i + 8] = b0hi[i];
            bf1[i] = b1lo[i]; bf1[i + 8] = b1hi[i];
        }

        acc0 = __builtin_amdgcn_wmma_f32_16x16x32_bf16(false, af, false, bf0, (short)0, acc0, false, false);
        acc1 = __builtin_amdgcn_wmma_f32_16x16x32_bf16(false, af, false, bf1, (short)0, acc1, false, false);
    }

    // D layout: lane holds column n = lr; VGPR r -> row m0 + r + half*8.
    // Fold b1 into ha only (sel==0).
    {
        const int h0 = n0 + lr;
        const int h1 = n0 + 16 + lr;
        const float bv0 = sel ? 0.0f : b1[h0];
        const float bv1 = sel ? 0.0f : b1[h1];
        #pragma unroll
        for (int r = 0; r < 8; ++r) {
            const int m = m0 + r + half * 8;
            C[(size_t)m * HH + h0] = acc0[r] + bv0;
            C[(size_t)m * HH + h1] = acc1[r] + bv1;
        }
    }
}

// ---------------------------------------------------------------------------
// Kernel 3: fused relu + W2 contraction via WMMA (N padded 2 -> 16).
// Block = (batch, t-tile of 16, 4 s-values). The 16x768 f32 hb tile (48 KB) is
// staged once into LDS (async-to-LDS when available), then shared by all
// 4 waves -> 4x less L2 read traffic. Row stride padded to 772 dwords so the
// 16 rows land in distinct bank groups (conflict-free ds_load_b128).
// ha already contains +b1 from the projection epilogue.
// ---------------------------------------------------------------------------
#define HB_STRIDE 772   // 768 + 4 dword pad

__global__ void scores_kernel(const float* __restrict__ ha,   // (B*S, H), bias folded in
                              const float* __restrict__ hb,   // (B*S, H)
                              const __bf16* __restrict__ w2t, // (16, H) padded W2^T
                              const float* __restrict__ b2,   // (O)
                              float* __restrict__ out)        // (B, S, S, O)
{
    __shared__ float hbT[16 * HB_STRIDE];     // 49408 B

    const int lane  = threadIdx.x & 31;
    const int widx  = threadIdx.x >> 5;       // wave in block: 0..3
    const int bid   = blockIdx.x;             // 1024 blocks
    const int bb    = bid >> 8;               // batch
    const int ttile = (bid >> 5) & 7;         // t-tile of 16
    const int sgrp  = bid & 31;               // s-group of 4
    const int s     = sgrp * 4 + widx;
    const int half  = lane >> 4;
    const int lr    = lane & 15;

    // ---- stage hb tile (16 rows x 768 f32) into LDS ----
    const float* hbg = hb + (size_t)(bb * SS + ttile * 16) * HH;
    #pragma unroll 4
    for (int i = 0; i < 24; ++i) {
        const int c   = threadIdx.x + 128 * i;   // float4 chunk id, 0..3071
        const int t   = c / 192;                 // row 0..15
        const int col = (c - t * 192) * 4;       // dword col 0..764
        const float* gsrc = hbg + (size_t)t * HH + col;
        float*       ldst = &hbT[t * HB_STRIDE + col];
#if HAVE_ASYNC_LDS
        __builtin_amdgcn_global_load_async_to_lds_b128(
            (gv4i_p)(size_t)gsrc,   // inttoptr into addrspace(1), drops const
            (lv4i_p)ldst,           // addrspacecast generic -> LDS
            0, 0);
#else
        *(float4*)ldst = *(const float4*)gsrc;
#endif
    }
#if HAVE_ASYNC_LDS
    wait_asynccnt0();
#endif
    __syncthreads();

    const float* harow = ha + (size_t)(bb * SS + s) * HH;
    const float* hbrow = &hbT[lr * HB_STRIDE];
    const __bf16* wrow = w2t + (size_t)lr * HH;

    v8f acc = {};

    for (int k0 = 0; k0 < HH; k0 += 32) {
        const int ka = k0 + half * 8;
        // 8+8 K-values per lane for the A fragment (relu(ha+hb) on the fly)
        const float4 A0 = *(const float4*)(harow + ka);
        const float4 A1 = *(const float4*)(harow + ka + 4);
        const float4 A2 = *(const float4*)(harow + ka + 16);
        const float4 A3 = *(const float4*)(harow + ka + 20);
        const float4 B0 = *(const float4*)(hbrow + ka);
        const float4 B1 = *(const float4*)(hbrow + ka + 4);
        const float4 B2 = *(const float4*)(hbrow + ka + 16);
        const float4 B3 = *(const float4*)(hbrow + ka + 20);

        v16bf af;
        af[0]  = relu_bf(A0.x + B0.x); af[1]  = relu_bf(A0.y + B0.y);
        af[2]  = relu_bf(A0.z + B0.z); af[3]  = relu_bf(A0.w + B0.w);
        af[4]  = relu_bf(A1.x + B1.x); af[5]  = relu_bf(A1.y + B1.y);
        af[6]  = relu_bf(A1.z + B1.z); af[7]  = relu_bf(A1.w + B1.w);
        af[8]  = relu_bf(A2.x + B2.x); af[9]  = relu_bf(A2.y + B2.y);
        af[10] = relu_bf(A2.z + B2.z); af[11] = relu_bf(A2.w + B2.w);
        af[12] = relu_bf(A3.x + B3.x); af[13] = relu_bf(A3.y + B3.y);
        af[14] = relu_bf(A3.z + B3.z); af[15] = relu_bf(A3.w + B3.w);

        const int kb = k0 + half * 8;
        v8bf wlo = *(const v8bf*)(wrow + kb);
        v8bf whi = *(const v8bf*)(wrow + kb + 16);
        v16bf bfw;
        #pragma unroll
        for (int i = 0; i < 8; ++i) { bfw[i] = wlo[i]; bfw[i + 8] = whi[i]; }

        acc = __builtin_amdgcn_wmma_f32_16x16x32_bf16(false, af, false, bfw, (short)0, acc, false, false);
    }

    // D: lane column n = lr (only n < OO meaningful), rows t = ttile*16 + r + half*8
    if (lr < OO) {
        const float bo = b2[lr];
        #pragma unroll
        for (int r = 0; r < 8; ++r) {
            const int t = ttile * 16 + r + half * 8;
            out[((size_t)(bb * SS + s) * SS + t) * OO + lr] = acc[r] + bo;
        }
    }
}

// ---------------------------------------------------------------------------
// Host launcher
// ---------------------------------------------------------------------------
extern "C" void kernel_launch(void* const* d_in, const int* in_sizes, int n_in,
                              void* d_out, int out_size, void* d_ws, size_t ws_size,
                              hipStream_t stream)
{
    (void)in_sizes; (void)n_in; (void)out_size; (void)ws_size;

    const float* a  = (const float*)d_in[0];
    const float* b  = (const float*)d_in[1];
    const float* W1 = (const float*)d_in[2];
    const float* b1 = (const float*)d_in[3];
    const float* W2 = (const float*)d_in[4];
    const float* b2 = (const float*)d_in[5];
    float* out = (float*)d_out;

    char* ws = (char*)d_ws;
    size_t off = 0;
    __bf16* a_bf  = (__bf16*)(ws + off); off += (size_t)BB * SS * DD * 2;   // 768 KB
    __bf16* b_bf  = (__bf16*)(ws + off); off += (size_t)BB * SS * DD * 2;   // 768 KB
    __bf16* w1aT  = (__bf16*)(ws + off); off += (size_t)HH * DD * 2;        // 1.125 MB
    __bf16* w1bT  = (__bf16*)(ws + off); off += (size_t)HH * DD * 2;        // 1.125 MB
    __bf16* w2t   = (__bf16*)(ws + off); off += (size_t)16 * HH * 2;        // 24 KB
    float*  haBuf = (float*)(ws + off);  off += (size_t)BB * SS * HH * 4;   // 1.5 MB
    float*  hbBuf = (float*)(ws + off);  off += (size_t)BB * SS * HH * 4;   // 1.5 MB

    // 1) convert / re-layout
    prep_kernel<<<512, 256, 0, stream>>>(a, b, W1, W2, a_bf, b_bf, w1aT, w1bT, w2t);

    // 2) both projections in one launch: 1536 waves (2 x 32 M-tiles x 24 N-tiles)
    proj_gemm_kernel<<<384, 128, 0, stream>>>(a_bf, b_bf, w1aT, w1bT, b1, haBuf, hbBuf);

    // 3) fused relu + W2: 1024 blocks = (4 batch x 8 t-tiles x 32 s-groups), 4 waves
    scores_kernel<<<1024, 128, 0, stream>>>(haBuf, hbBuf, w2t, b2, out);
}
